// AttentionLayer_86337432584605
// MI455X (gfx1250) — matrix operand order repaired and checked by reading, not compiled, not run on hardware
//
#include <hip/hip_runtime.h>
#include <hip/hip_bf16.h>
#include <math.h>

// Bahdanau additive attention for MI455X (gfx1250, wave32, WMMA).
// B=32, T=2048, D=1024. Dominant op: h@U_a (137 GFLOP) fused with tanh+v-dot.
#define BB   32
#define TT   2048
#define DD   1024
#define MROWS (BB*TT)          // 65536 fused (b,t) rows
#define MBLK 128               // rows per block (one b each: 128 | 2048)

typedef __attribute__((ext_vector_type(16))) __bf16 v16bf;
typedef __attribute__((ext_vector_type(8)))  __bf16 v8bf;
typedef __attribute__((ext_vector_type(4)))  __bf16 v4bf;
typedef __attribute__((ext_vector_type(8)))  float  v8f;

// Hardware tanh (CDNA5 TRANS op). Builtin if declared, else raw v_tanh_f32.
__device__ __forceinline__ float hw_tanh(float x) {
#if __has_builtin(__builtin_amdgcn_tanhf)
    return __builtin_amdgcn_tanhf(x);
#else
    float r;
    asm volatile("v_tanh_f32 %0, %1\n\tv_nop" : "=v"(r) : "v"(x));
    return r;
#endif
}

// ---------------------------------------------------------------------------
// Kernel A: wsb[b][o] = sum_i s[b][i] * W_a[i][o]   (negligible: 67 MFLOP)
// ---------------------------------------------------------------------------
__global__ __launch_bounds__(256) void ws_kernel(const float* __restrict__ s,
                                                 const float* __restrict__ W,
                                                 float* __restrict__ wsb) {
    const int o = blockIdx.x * 256 + threadIdx.x;
    const int b = blockIdx.y;
    const float* sr = s + b * DD;
    float acc = 0.f;
    for (int i = 0; i < DD; ++i)
        acc = fmaf(sr[i], W[i * DD + o], acc);
    wsb[b * DD + o] = acc;
}

// ---------------------------------------------------------------------------
// Main WMMA kernel helpers (512 threads = 16 wave32s per block)
// ---------------------------------------------------------------------------
// B-tile stage: thread owns 2 columns x 4 contiguous K (32x128 tile, 8KB).
// Loads: float2, coalesced across lanes. Stores: two 8B ds_store_b64.
__device__ __forceinline__ void loadB(const float* __restrict__ U, int k0,
                                      int nChunk, int tid, float2* f) {
    const int n2 = (tid & 63) * 2;
    const int kg = tid >> 6;                                  // 0..7
    const float2* U2 = (const float2*)(U + (size_t)(k0 + kg * 4) * DD + nChunk + n2);
    #pragma unroll
    for (int i = 0; i < 4; ++i)
        f[i] = U2[(size_t)i * (DD / 2)];                      // row stride 1024 floats
}
__device__ __forceinline__ void storeB(__bf16* __restrict__ buf, int tid,
                                       const float2* f) {
    const int n2 = (tid & 63) * 2;
    const int kg = tid >> 6;
    v4bf p0, p1;
    #pragma unroll
    for (int i = 0; i < 4; ++i) { p0[i] = (__bf16)f[i].x; p1[i] = (__bf16)f[i].y; }
    *(v4bf*)(buf + (n2 + 0) * 32 + kg * 4) = p0;              // [n][k] layout
    *(v4bf*)(buf + (n2 + 1) * 32 + kg * 4) = p1;
}

// One K-step of WMMAs: load ALL fragments into distinct registers first so
// the ds_load_b128s pipeline against the WMMA chain, then issue 4 WMMAs.
__device__ __forceinline__ void wmma_step(const __bf16* __restrict__ lds_a,
                                          const __bf16* __restrict__ bbuf,
                                          int aRow, int k0, int nw, int lc,
                                          int half, v8f acc[4]) {
    // A 16x32 bf16 layout: lanes 0-15 K 0..7|16..23, lanes 16-31 K 8..15|24..31
    const __bf16* ap = lds_a + aRow * DD + k0;
    v8bf alo = *(const v8bf*)(ap + 8 * half);
    v8bf ahi = *(const v8bf*)(ap + 16 + 8 * half);
    v16bf af = __builtin_shufflevector(alo, ahi,
                 0,1,2,3,4,5,6,7,8,9,10,11,12,13,14,15);
    v16bf bv[4];
    #pragma unroll
    for (int j = 0; j < 4; ++j) {
        // B 32x16: lane = col; lanes 0-15 K 0..7|16..23, lanes 16-31 K 8..15|24..31
        const __bf16* bp = bbuf + (nw * 64 + j * 16 + lc) * 32;
        v8bf blo = *(const v8bf*)(bp + 8 * half);
        v8bf bhi = *(const v8bf*)(bp + 16 + 8 * half);
        bv[j] = __builtin_shufflevector(blo, bhi,
                  0,1,2,3,4,5,6,7,8,9,10,11,12,13,14,15);
    }
    #pragma unroll
    for (int j = 0; j < 4; ++j)
        acc[j] = __builtin_amdgcn_wmma_f32_16x16x32_bf16(
                     false, af, false, bv[j], (short)0, acc[j], false, false);
}

// ---------------------------------------------------------------------------
// Kernel B: e[m] = v . tanh(wsb[b] + h[m] @ U_a), bf16 WMMA, fused epilogue.
//   512 blocks x 512 threads (16 wave32: 8 M-waves x 2 N-waves).
//   LDS (272.5 KB of the 320 KB WGP pool):
//     A tile 128x1024 bf16 resident -> h read from HBM exactly once,
//     B tile double-buffered 2 x (128x32) bf16 -> U_a L2 traffic ~2 GB total,
//     red[128] for the row reduction.
// ---------------------------------------------------------------------------
__global__ __launch_bounds__(512) void e_kernel(const float* __restrict__ h,
                                                const float* __restrict__ U,
                                                const float* __restrict__ wsb,
                                                const float* __restrict__ va,
                                                float* __restrict__ e) {
    extern __shared__ __align__(16) char smem[];
    __bf16* lds_a  = (__bf16*)smem;                                // 256 KB
    __bf16* lds_b0 = (__bf16*)(smem + MBLK * DD * 2);              // 8 KB
    __bf16* lds_b1 = (__bf16*)(smem + MBLK * DD * 2 + 8192);       // 8 KB
    float*  red    = (float*)(smem + MBLK * DD * 2 + 16384);       // 512 B

    const int tid   = threadIdx.x;
    const int mBase = blockIdx.x * MBLK;
    const int bIdx  = mBase >> 11;

    // ---- stage A: 128 rows x 1024 K of h -> bf16, row-major ----
    {
        const float4* h4 = (const float4*)(h + (size_t)mBase * DD);
        for (int i = tid; i < MBLK * DD / 4; i += 512) {
            float4 f = h4[i];
            __bf16* p = lds_a + i * 4;
            p[0] = (__bf16)f.x; p[1] = (__bf16)f.y;
            p[2] = (__bf16)f.z; p[3] = (__bf16)f.w;
        }
    }

    const int lane = tid & 31;
    const int wave = tid >> 5;        // 0..15
    const int mw   = wave >> 1;       // 0..7 : rows mw*16 .. +15
    const int nw   = wave & 1;        // 0..1 : cols nw*64 .. +63 of chunk
    const int half = lane >> 4;
    const int lc   = lane & 15;
    const int aRow = mw * 16 + lc;

    float partial[8];
    #pragma unroll
    for (int r = 0; r < 8; ++r) partial[r] = 0.f;

    for (int chunk = 0; chunk < 8; ++chunk) {
        const int nChunk = chunk * 128;
        v8f acc[4] = {};

        // prologue: stage K-step 0 into buffer 0
        {
            float2 f[4];
            loadB(U, 0, nChunk, tid, f);
            storeB(lds_b0, tid, f);
        }

        for (int kk = 0; kk < 32; kk += 2) {
            // even step: compute from b0, prefetch kk+1 into b1
            {
                __syncthreads();
                float2 f[4];
                loadB(U, (kk + 1) * 32, nChunk, tid, f);     // issue early
                wmma_step(lds_a, lds_b0, aRow, kk * 32, nw, lc, half, acc);
                storeB(lds_b1, tid, f);
            }
            // odd step: compute from b1, prefetch kk+2 into b0
            {
                __syncthreads();
                const bool pf = (kk + 2) < 32;
                float2 f[4];
                if (pf) loadB(U, (kk + 2) * 32, nChunk, tid, f);
                wmma_step(lds_a, lds_b1, aRow, (kk + 1) * 32, nw, lc, half, acc);
                if (pf) storeB(lds_b0, tid, f);
            }
        }

        // fused epilogue: +bias, hw tanh, dot with v_a
        // C layout: (VGPR r, lane) -> M = r + 8*half, N = lc
        const int colBase = nChunk + nw * 64;
        #pragma unroll
        for (int j = 0; j < 4; ++j) {
            const int n  = colBase + j * 16 + lc;
            const float vn = va[n];
            const float wn = wsb[bIdx * DD + n];
            #pragma unroll
            for (int r = 0; r < 8; ++r)
                partial[r] += vn * hw_tanh(acc[j][r] + wn);
        }
    }

    // reduce over 16 cols per tile, then the 2 N-waves, via LDS
    __syncthreads();
    if (tid < MBLK) red[tid] = 0.f;
    __syncthreads();
    #pragma unroll
    for (int r = 0; r < 8; ++r) {
        float p = partial[r];
        for (int off = 8; off >= 1; off >>= 1)
            p += __shfl_xor(p, off, 16);
        if (lc == 0)
            atomicAdd(&red[mw * 16 + half * 8 + r], p);      // ds_add_f32
    }
    __syncthreads();
    if (tid < MBLK) e[mBase + tid] = red[tid];
}

// ---------------------------------------------------------------------------
// Kernel C1: softmax over T per batch row, in place (e -> a).
// ---------------------------------------------------------------------------
__global__ __launch_bounds__(256) void softmax_kernel(float* __restrict__ e) {
    __shared__ float sred[256];
    const int b = blockIdx.x, tid = threadIdx.x;
    float* er = e + b * TT;

    float m = -3.4e38f;
    for (int t = tid; t < TT; t += 256) m = fmaxf(m, er[t]);
    sred[tid] = m; __syncthreads();
    for (int s2 = 128; s2 > 0; s2 >>= 1) {
        if (tid < s2) sred[tid] = fmaxf(sred[tid], sred[tid + s2]);
        __syncthreads();
    }
    m = sred[0]; __syncthreads();

    float sum = 0.f;
    for (int t = tid; t < TT; t += 256) sum += __expf(er[t] - m);
    sred[tid] = sum; __syncthreads();
    for (int s2 = 128; s2 > 0; s2 >>= 1) {
        if (tid < s2) sred[tid] += sred[tid + s2];
        __syncthreads();
    }
    const float inv = 1.f / sred[0];

    for (int t = tid; t < TT; t += 256)
        er[t] = __expf(er[t] - m) * inv;
}

// ---------------------------------------------------------------------------
// Kernel C2: c[b][d] = sum_t a[b][t]*h[b][t][d], T split x4 for parallelism,
// deterministic two-stage reduction (no fp atomics -> replay bit-identical).
// ---------------------------------------------------------------------------
__global__ __launch_bounds__(256) void ctx_part_kernel(const float* __restrict__ a,
                                                       const float* __restrict__ h,
                                                       float* __restrict__ cpart) {
    const int d  = blockIdx.x * 256 + threadIdx.x;   // 0..1023
    const int b  = blockIdx.y;                       // 0..31
    const int tz = blockIdx.z;                       // 0..3
    const float* ab = a + b * TT + tz * 512;
    const float* hb = h + ((size_t)b * TT + tz * 512) * DD + d;
    float acc = 0.f;
    for (int t = 0; t < 512; ++t)
        acc = fmaf(ab[t], hb[(size_t)t * DD], acc);
    cpart[(size_t)tz * BB * DD + b * DD + d] = acc;
}

__global__ __launch_bounds__(256) void ctx_reduce_kernel(const float* __restrict__ cpart,
                                                         float* __restrict__ out) {
    const int i = blockIdx.x * 256 + threadIdx.x;    // 0..32767
    out[i] = (cpart[i]                 + cpart[(size_t)BB * DD + i]) +
             (cpart[(size_t)2*BB*DD+i] + cpart[(size_t)3*BB*DD + i]);
}

// ---------------------------------------------------------------------------
extern "C" void kernel_launch(void* const* d_in, const int* in_sizes, int n_in,
                              void* d_out, int out_size, void* d_ws, size_t ws_size,
                              hipStream_t stream) {
    (void)in_sizes; (void)n_in; (void)out_size; (void)ws_size;
    const float* s  = (const float*)d_in[0];   // [32,1024]
    const float* h  = (const float*)d_in[1];   // [32,2048,1024]
    const float* W  = (const float*)d_in[2];   // [1024,1024]
    const float* U  = (const float*)d_in[3];   // [1024,1024]
    const float* va = (const float*)d_in[4];   // [1024]
    float* out = (float*)d_out;                // [32,1024]

    // scratch layout: e/a [65536]f @0 | wsb [32768]f @256KB | cpart [4*32768]f @384KB
    float* e     = (float*)d_ws;
    float* wsb   = (float*)((char*)d_ws + (size_t)MROWS * 4);
    float* cpart = (float*)((char*)d_ws + (size_t)MROWS * 4 + (size_t)BB * DD * 4);

    ws_kernel<<<dim3(4, 32), 256, 0, stream>>>(s, W, wsb);

    const size_t shmem = (size_t)MBLK * DD * 2 + 2 * 8192 + MBLK * 4;  // ~272.5 KB
    e_kernel<<<dim3(MROWS / MBLK), 512, shmem, stream>>>(h, U, wsb, va, e);

    softmax_kernel<<<dim3(32), 256, 0, stream>>>(e);

    ctx_part_kernel<<<dim3(4, 32, 4), 256, 0, stream>>>(e, h, cpart);
    ctx_reduce_kernel<<<dim3(128), 256, 0, stream>>>(cpart, out);
}
